// MAGIC_21423296873027
// MI455X (gfx1250) — matrix operand order repaired and checked by reading, not compiled, not run on hardware
//
#include <hip/hip_runtime.h>
#include <hip/hip_bf16.h>
#include <math.h>

#define NN 256

typedef __attribute__((ext_vector_type(16))) _Float16 v16h;
typedef __attribute__((ext_vector_type(8)))  _Float16 v8h;
typedef __attribute__((ext_vector_type(8)))  float    v8f;
typedef __attribute__((ext_vector_type(4)))  int      v4i;

#if defined(__gfx1250__) && __has_builtin(__builtin_amdgcn_global_load_async_to_lds_b128)
#define HAVE_ASYNC 1
#else
#define HAVE_ASYNC 0
#endif

// ---------------------------------------------------------------- WMMA helpers
__device__ __forceinline__ v8f wmma16(v16h a, v16h b, v8f c) {
  return __builtin_amdgcn_wmma_f32_16x16x32_f16(false, a, false, b, (short)0, c,
                                                false, false);
}

// A fragment: 16x32 f16 tile, row-major LDS, stride ld halves (ld % 8 == 0).
// lanes 0-15: M=lane, elems 0..7=K0..7, 8..15=K16..23; lanes 16-31: K+8.
__device__ __forceinline__ v16h frag_a16(const _Float16* p, int ld) {
  int lane = threadIdx.x & 31;
  int m  = lane & 15;
  int kb = (lane >> 4) << 3;
  const v8h* q0 = (const v8h*)(p + m * ld + kb);
  const v8h* q1 = (const v8h*)(p + m * ld + kb + 16);
  v8h lo = *q0, hi = *q1;
  return __builtin_shufflevector(lo, hi, 0, 1, 2, 3, 4, 5, 6, 7, 8, 9, 10, 11,
                                 12, 13, 14, 15);
}

// B fragment from TRANSPOSED B tile (rows = output col n, cols = K).
// elems 0..7 = BT[n][kb+0..7], 8..15 = BT[n][kb+16..23], kb = 8*(lane/16).
__device__ __forceinline__ v16h frag_bt16(const _Float16* p, int ld) {
  int lane = threadIdx.x & 31;
  int n  = lane & 15;
  int kb = (lane >> 4) << 3;
  const v8h* q0 = (const v8h*)(p + n * ld + kb);
  const v8h* q1 = (const v8h*)(p + n * ld + kb + 16);
  v8h lo = *q0, hi = *q1;
  return __builtin_shufflevector(lo, hi, 0, 1, 2, 3, 4, 5, 6, 7, 8, 9, 10, 11,
                                 12, 13, 14, 15);
}

// 16-byte global->LDS copy: async gfx1250 path (ASYNCcnt), vector fallback.
__device__ __forceinline__ void cp16(const _Float16* g, _Float16* l) {
#if HAVE_ASYNC
  typedef __attribute__((address_space(1))) v4i* g4p;
  typedef __attribute__((address_space(3))) v4i* l4p;
  __builtin_amdgcn_global_load_async_to_lds_b128(
      (g4p)(unsigned long long)(const void*)g,
      (l4p)(unsigned)(unsigned long long)(const void*)l, 0, 0);
#else
  *(v8h*)l = *(const v8h*)g;
#endif
}
__device__ __forceinline__ void cp16_wait() {
#if HAVE_ASYNC
#if __has_builtin(__builtin_amdgcn_s_wait_asynccnt)
  __builtin_amdgcn_s_wait_asynccnt(0);
#else
  asm volatile("s_wait_asynccnt 0x0" ::: "memory");
#endif
#endif
}

// ---------------------------------------------------------------- generic GEMM
// C[M,N] (+)= A[M,K](f16) @ B[K,N] (+ bias).  B passed TRANSPOSED: BT[N,K].
// K must be a multiple of 32 (true for all call sites).
template <bool ACCUM, bool BIAS>
__global__ __launch_bounds__(256) void gemm_wmma(
    const _Float16* __restrict__ A, int lda, long sA,
    const _Float16* __restrict__ BT, int ldb, long sB,
    const float* __restrict__ bias,
    float* __restrict__ C, int ldc, long sC,
    int M, int N, int K) {
  __shared__ __attribute__((aligned(16))) _Float16 As[64][40];
  __shared__ __attribute__((aligned(16))) _Float16 Bs[64][40];
  A  += (long)blockIdx.z * sA;
  BT += (long)blockIdx.z * sB;
  C  += (long)blockIdx.z * sC;
  const int m0 = blockIdx.y * 64, n0 = blockIdx.x * 64;
  const int tid = threadIdx.x;
  const int wave = tid >> 5, lane = tid & 31;
  const int tm  = wave & 3;
  const int tn0 = (wave >> 2) * 2;
  const int nl = lane & 15, mb = (lane >> 4) * 8;

  // staging assignment: thread -> (row, 8-half chunk)
  const int sr = tid >> 2;
  const int sc = (tid & 3) << 3;
  int gm = m0 + sr; gm = gm < M ? gm : M - 1;   // clamp: OOB rows only feed
  int gn = n0 + sr; gn = gn < N ? gn : N - 1;   // never-stored outputs

  v8f acc0{}, acc1{};
  if (ACCUM) {
#pragma unroll
    for (int r = 0; r < 8; ++r) {
      int rm = m0 + tm * 16 + mb + r;
      int g0 = n0 + tn0 * 16 + nl, g1 = g0 + 16;
      if (rm < M) {
        if (g0 < N) acc0[r] = C[(long)rm * ldc + g0];
        if (g1 < N) acc1[r] = C[(long)rm * ldc + g1];
      }
    }
  }

  for (int k0 = 0; k0 < K; k0 += 32) {
    cp16(A  + (long)gm * lda + k0 + sc, &As[sr][sc]);
    cp16(BT + (long)gn * ldb + k0 + sc, &Bs[sr][sc]);
    cp16_wait();
    __syncthreads();
    v16h af = frag_a16(&As[tm * 16][0], 40);
    v16h b0 = frag_bt16(&Bs[tn0 * 16][0], 40);
    v16h b1 = frag_bt16(&Bs[tn0 * 16 + 16][0], 40);
    acc0 = wmma16(af, b0, acc0);
    acc1 = wmma16(af, b1, acc1);
    __syncthreads();
  }

#pragma unroll
  for (int r = 0; r < 8; ++r) {
    int rm = m0 + tm * 16 + mb + r;
    if (rm >= M) continue;
    int g0 = n0 + tn0 * 16 + nl, g1 = g0 + 16;
    if (g0 < N) C[(long)rm * ldc + g0] = acc0[r] + (BIAS ? bias[g0] : 0.f);
    if (g1 < N) C[(long)rm * ldc + g1] = acc1[r] + (BIAS ? bias[g1] : 0.f);
  }
}

// ---------------------------------------------------- pairwise hard-adjacency
__device__ __forceinline__ float hash_uniform(unsigned s, unsigned i,
                                              unsigned j, unsigned c) {
  unsigned long long x = ((unsigned long long)s << 48) ^
                         ((unsigned long long)i << 28) ^
                         ((unsigned long long)j << 8) ^ c;
  x += 0x9E3779B97F4A7C15ull;
  x ^= x >> 30; x *= 0xBF58476D1CE4E5B9ull;
  x ^= x >> 27; x *= 0x94D049BB133111EBull;
  x ^= x >> 31;
  float u = (float)(unsigned)(x >> 40) * (1.f / 16777216.f);
  return u < 1e-9f ? 1e-9f : u;
}

// Block = 8x8 tile of (i,j) pairs.  X[64,256]=relu(li[i]+lj[j]+b1) in LDS f16,
// WMMA X@w2 (w2 pre-transposed f16) -> relu -> Y[64,64] -> 2 logits ->
// Gumbel argmax -> adj in {0,1}.
__global__ __launch_bounds__(256) void hard_adj_kernel(
    const float* __restrict__ li, const float* __restrict__ lj,
    const float* __restrict__ b1, const _Float16* __restrict__ w2T,
    const float* __restrict__ b2, const float* __restrict__ w3,
    const float* __restrict__ b3, float* __restrict__ adj, unsigned seed) {
  __shared__ __attribute__((aligned(16))) _Float16 Xs[64][264];
  __shared__ __attribute__((aligned(16))) _Float16 Ws[64][40];
  __shared__ float Ys[64][66];
  const int i0 = blockIdx.y * 8, j0 = blockIdx.x * 8;
  const int tid = threadIdx.x;

  for (int ch = tid; ch < 64 * 32; ch += 256) {
    int p = ch >> 5, k8 = (ch & 31) << 3;
    int gi = i0 + (p >> 3), gj = j0 + (p & 7);
    const float* lip = li + gi * 256 + k8;
    const float* ljp = lj + gj * 256 + k8;
    v8h x;
#pragma unroll
    for (int t = 0; t < 8; ++t) {
      float v = lip[t] + ljp[t] + b1[k8 + t];
      x[t] = (_Float16)(v > 0.f ? v : 0.f);
    }
    *(v8h*)&Xs[p][k8] = x;
  }

  const int wave = tid >> 5, lane = tid & 31;
  const int tm = wave & 3, tn0 = (wave >> 2) * 2;
  const int nl = lane & 15, mb = (lane >> 4) * 8;
  const int sr = tid >> 2, sc = (tid & 3) << 3;
  v8f acc0{}, acc1{};
  for (int k0 = 0; k0 < 256; k0 += 32) {
    cp16(w2T + sr * 256 + k0 + sc, &Ws[sr][sc]);  // w2T: 64 x 256
    cp16_wait();
    __syncthreads();
    v16h af  = frag_a16(&Xs[tm * 16][k0], 264);
    v16h bf0 = frag_bt16(&Ws[tn0 * 16][0], 40);
    v16h bf1 = frag_bt16(&Ws[tn0 * 16 + 16][0], 40);
    acc0 = wmma16(af, bf0, acc0);
    acc1 = wmma16(af, bf1, acc1);
    __syncthreads();
  }
#pragma unroll
  for (int r = 0; r < 8; ++r) {
    int pm = tm * 16 + mb + r;
    float y0 = acc0[r] + b2[tn0 * 16 + nl];
    float y1 = acc1[r] + b2[tn0 * 16 + 16 + nl];
    Ys[pm][tn0 * 16 + nl]      = y0 > 0.f ? y0 : 0.f;
    Ys[pm][tn0 * 16 + 16 + nl] = y1 > 0.f ? y1 : 0.f;
  }
  __syncthreads();
  if (tid < 64) {
    int gi = i0 + (tid >> 3), gj = j0 + (tid & 7);
    float l0 = b3[0], l1 = b3[1];
    for (int k = 0; k < 64; ++k) {
      float y = Ys[tid][k];
      l0 += y * w3[k * 2 + 0];
      l1 += y * w3[k * 2 + 1];
    }
    float g0 = -logf(-logf(hash_uniform(seed, gi, gj, 0)));
    float g1 = -logf(-logf(hash_uniform(seed, gi, gj, 1)));
    adj[gi * NN + gj] = (l1 + g1 > l0 + g0) ? 1.f : 0.f;
  }
}

// ---------------------------------------------------------------- conversions
__global__ void conv_kernel(const float* __restrict__ in,
                            _Float16* __restrict__ out, int n) {
  int idx = blockIdx.x * 256 + threadIdx.x;
  if (idx < n) out[idx] = (_Float16)in[idx];
}
// in: K x N row-major f32  ->  out: N x K row-major f16 (transposed)
__global__ void conv_t_kernel(const float* __restrict__ in,
                              _Float16* __restrict__ out, int K, int N) {
  int idx = blockIdx.x * 256 + threadIdx.x;
  if (idx >= K * N) return;
  int k = idx / N, n = idx % N;
  out[(long)n * K + k] = (_Float16)in[idx];
}

// ---------------------------------------------------------------- small kernels
__global__ void lstm_kernel(const float* __restrict__ gates,
                            const float* __restrict__ cell,
                            float* __restrict__ hout, float* __restrict__ cout,
                            float* __restrict__ hws, _Float16* __restrict__ h16) {
  int idx = blockIdx.x * 256 + threadIdx.x;  // 256*512
  int r = idx >> 9, c = idx & 511;
  const float* g = gates + (long)r * 2048;
  float gi = g[c], gf = g[512 + c], gg = g[1024 + c], go = g[1536 + c];
  float si = 1.f / (1.f + expf(-gi));
  float sf = 1.f / (1.f + expf(-gf));
  float so = 1.f / (1.f + expf(-go));
  float cn = sf * cell[idx] + si * tanhf(gg);
  float hn = so * tanhf(cn);
  hout[idx] = hn; cout[idx] = cn; hws[idx] = hn; h16[idx] = (_Float16)hn;
}

__global__ void gat_coef_kernel(const float* __restrict__ hproj,
                                const float* __restrict__ ai,
                                const float* __restrict__ aj,
                                float* __restrict__ ci, float* __restrict__ cj,
                                int H, int F) {
  __shared__ float red[128];
  int n = blockIdx.x, t = threadIdx.x;
  for (int h = 0; h < H; ++h) {
    float si = 0.f, sj = 0.f;
    for (int f = t; f < F; f += 128) {
      float hv = hproj[(long)n * H * F + h * F + f];
      si += hv * ai[h * F + f];
      sj += hv * aj[h * F + f];
    }
    red[t] = si; __syncthreads();
    for (int s = 64; s > 0; s >>= 1) { if (t < s) red[t] += red[t + s]; __syncthreads(); }
    if (t == 0) ci[n * H + h] = red[0];
    __syncthreads();
    red[t] = sj; __syncthreads();
    for (int s = 64; s > 0; s >>= 1) { if (t < s) red[t] += red[t + s]; __syncthreads(); }
    if (t == 0) cj[n * H + h] = red[0];
    __syncthreads();
  }
}

// att[h][i][j] (f16): softmax_j over neighbors (forced self-loop), 0 elsewhere.
__global__ void gat_att_kernel(const float* __restrict__ ci,
                               const float* __restrict__ cj,
                               const float* __restrict__ adj,
                               _Float16* __restrict__ att, int H) {
  __shared__ float buf[256];
  int i = blockIdx.x, h = blockIdx.y, j = threadIdx.x;
  float e = ci[i * H + h] + cj[j * H + h];
  e = e > 0.f ? e : 0.2f * e;  // leaky_relu NEG=0.2
  bool nb = (i == j) || (adj[i * NN + j] > 0.f);
  buf[j] = nb ? e : -1e30f;
  __syncthreads();
  for (int s = 128; s > 0; s >>= 1) { if (j < s) buf[j] = fmaxf(buf[j], buf[j + s]); __syncthreads(); }
  float mx = buf[0];
  __syncthreads();
  float ex = nb ? expf(e - mx) : 0.f;
  buf[j] = ex;
  __syncthreads();
  for (int s = 128; s > 0; s >>= 1) { if (j < s) buf[j] += buf[j + s]; __syncthreads(); }
  att[(long)h * NN * NN + (long)i * NN + j] = (_Float16)(ex / buf[0]);
}

__global__ void bias_elu_kernel(const float* __restrict__ in,
                                const float* __restrict__ b,
                                float* __restrict__ out,
                                _Float16* __restrict__ out16) {
  int idx = blockIdx.x * 256 + threadIdx.x;
  float v = in[idx] + b[idx & 511];
  v = v > 0.f ? v : expf(v) - 1.f;
  out[idx] = v; out16[idx] = (_Float16)v;
}

__global__ void bias_add_kernel(const float* __restrict__ in,
                                const float* __restrict__ b,
                                float* __restrict__ out,
                                _Float16* __restrict__ out16) {
  int idx = blockIdx.x * 256 + threadIdx.x;
  float v = in[idx] + b[idx & 511];
  out[idx] = v; out16[idx] = (_Float16)v;
}

__global__ void cat_kernel(const float* __restrict__ a,
                           const float* __restrict__ b,
                           float* __restrict__ out,
                           _Float16* __restrict__ out16) {
  int idx = blockIdx.x * 256 + threadIdx.x;  // 256*1024
  int r = idx >> 10, c = idx & 1023;
  float v = (c < 512) ? a[r * 512 + c] : b[r * 512 + c - 512];
  out[idx] = v; out16[idx] = (_Float16)v;
}

__global__ void value_kernel(const float* __restrict__ catb,
                             const float* __restrict__ vw,
                             const float* __restrict__ vb,
                             float* __restrict__ out) {
  __shared__ float red[128];
  int r = blockIdx.x, t = threadIdx.x;
  float s = 0.f;
  for (int k = t; k < 1024; k += 128) s += catb[(long)r * 1024 + k] * vw[k];
  red[t] = s; __syncthreads();
  for (int st = 64; st > 0; st >>= 1) { if (t < st) red[t] += red[t + st]; __syncthreads(); }
  if (t == 0) out[r] = red[0] + vb[0];
}

__global__ void logsoftmax_kernel(const float* __restrict__ logits,
                                  float* __restrict__ out) {
  __shared__ float buf[32];
  int r = blockIdx.x, t = threadIdx.x;  // block 32
  float x = (t < 20) ? logits[r * 20 + t] : -1e30f;
  buf[t] = x; __syncthreads();
  for (int s = 16; s > 0; s >>= 1) { if (t < s) buf[t] = fmaxf(buf[t], buf[t + s]); __syncthreads(); }
  float mx = buf[0]; __syncthreads();
  float ex = (t < 20) ? expf(x - mx) : 0.f;
  buf[t] = ex; __syncthreads();
  for (int s = 16; s > 0; s >>= 1) { if (t < s) buf[t] += buf[t + s]; __syncthreads(); }
  float lse = logf(buf[0]) + mx;
  if (t < 20) out[r * 20 + t] = x - lse;
}

// ---------------------------------------------------------------- launcher
extern "C" void kernel_launch(void* const* d_in, const int* in_sizes, int n_in,
                              void* d_out, int out_size, void* d_ws,
                              size_t ws_size, hipStream_t stream) {
  (void)in_sizes; (void)n_in; (void)out_size; (void)ws_size;
  const float* obs    = (const float*)d_in[0];
  const float* hidden = (const float*)d_in[1];
  const float* cell   = (const float*)d_in[2];
  const float* enc_w  = (const float*)d_in[3];
  const float* enc_b  = (const float*)d_in[4];
  const float* wih    = (const float*)d_in[5];
  const float* whh    = (const float*)d_in[6];
  const float* bih    = (const float*)d_in[7];
  const float* bhh    = (const float*)d_in[8];
  const float* menc_w = (const float*)d_in[9];
  const float* menc_b = (const float*)d_in[10];
  const float* h1_w1  = (const float*)d_in[11];
  const float* h1_b1  = (const float*)d_in[12];
  const float* h1_w2  = (const float*)d_in[13];
  const float* h1_b2  = (const float*)d_in[14];
  const float* h1_w3  = (const float*)d_in[15];
  const float* h1_b3  = (const float*)d_in[16];
  const float* h2_w1  = (const float*)d_in[17];
  const float* h2_b1  = (const float*)d_in[18];
  const float* h2_w2  = (const float*)d_in[19];
  const float* h2_b2  = (const float*)d_in[20];
  const float* h2_w3  = (const float*)d_in[21];
  const float* h2_b3  = (const float*)d_in[22];
  const float* g1_W   = (const float*)d_in[23];
  const float* g1_ai  = (const float*)d_in[24];
  const float* g1_aj  = (const float*)d_in[25];
  const float* g1_b   = (const float*)d_in[26];
  const float* g2_W   = (const float*)d_in[27];
  const float* g2_ai  = (const float*)d_in[28];
  const float* g2_aj  = (const float*)d_in[29];
  const float* g2_b   = (const float*)d_in[30];
  const float* mdec_w = (const float*)d_in[31];
  const float* mdec_b = (const float*)d_in[32];
  const float* val_w  = (const float*)d_in[33];
  const float* val_b  = (const float*)d_in[34];
  const float* act_w  = (const float*)d_in[35];
  const float* act_b  = (const float*)d_in[36];

  // ---------------- f32 workspace ----------------
  float* W = (float*)d_ws;
  float* x_enc  = W;               // 256*512
  float* gates  = x_enc + 131072;  // 256*2048
  float* h_new  = gates + 524288;
  float* comm0  = h_new + 131072;
  float* li1    = comm0 + 131072;  // 256*256 x4
  float* lj1    = li1 + 65536;
  float* li2    = lj1 + 65536;
  float* lj2    = li2 + 65536;
  float* adj1   = lj2 + 65536;
  float* adj2   = adj1 + 65536;
  float* hproj1 = adj2 + 65536;    // 256*512
  float* ci1    = hproj1 + 131072; // 256*4
  float* cj1    = ci1 + 1024;
  float* g1o    = cj1 + 1024;      // 256*512
  float* comm2  = g1o + 131072;
  float* hproj2 = comm2 + 131072;
  float* ci2    = hproj2 + 131072; // 256
  float* cj2    = ci2 + 256;
  float* g2o    = cj2 + 256;
  float* comm3  = g2o + 131072;
  float* comm4  = comm3 + 131072;
  float* catb   = comm4 + 131072;  // 256*1024
  float* logits = catb + 262144;   // 256*20 (padded)
  float* f32end = logits + 5376;

  // ---------------- f16 workspace ----------------
  _Float16* Hb = (_Float16*)f32end;
  _Float16* obs16   = Hb;              // 256*128
  _Float16* hid16   = obs16 + 32768;   // 256*512
  _Float16* x16     = hid16 + 131072;
  _Float16* h16     = x16 + 131072;
  _Float16* c016    = h16 + 131072;
  _Float16* encT    = c016 + 131072;   // 512*128
  _Float16* wihT    = encT + 65536;    // 2048*512
  _Float16* whhT    = wihT + 1048576;
  _Float16* mencT   = whhT + 1048576;  // 512*512
  _Float16* w1aT1   = mencT + 262144;  // 256*512 x4
  _Float16* w1bT1   = w1aT1 + 131072;
  _Float16* w1aT2   = w1bT1 + 131072;
  _Float16* w1bT2   = w1aT2 + 131072;
  _Float16* w2T1    = w1bT2 + 131072;  // 64*256
  _Float16* w2T2    = w2T1 + 16384;
  _Float16* g1WT    = w2T2 + 16384;    // 512*512
  _Float16* g2WT    = g1WT + 262144;
  _Float16* mdecT   = g2WT + 262144;
  _Float16* actWT   = mdecT + 262144;  // 20*1024
  _Float16* hp1T    = actWT + 20480;   // 512*256
  _Float16* hp2T    = hp1T + 131072;   // 512*256
  _Float16* att116  = hp2T + 131072;   // 4*256*256
  _Float16* att216  = att116 + 262144; // 256*256
  _Float16* c216    = att216 + 65536;  // 256*512
  _Float16* c316    = c216 + 131072;
  _Float16* cat16   = c316 + 131072;   // 256*1024

  float* out_action = (float*)d_out;      // 1*256*20
  float* out_value  = out_action + 5120;  // 256
  float* out_h      = out_value + 256;    // 256*512
  float* out_c      = out_h + 131072;     // 256*512

  dim3 blk(256);
#define GRID1(n) dim3(((n) + 255) / 256)

  // ---- one-time f16 conversions (weights transposed for BT staging) ----
  conv_kernel<<<GRID1(32768), blk, 0, stream>>>(obs, obs16, 32768);
  conv_kernel<<<GRID1(131072), blk, 0, stream>>>(hidden, hid16, 131072);
  conv_t_kernel<<<GRID1(65536), blk, 0, stream>>>(enc_w, encT, 128, 512);
  conv_t_kernel<<<GRID1(1048576), blk, 0, stream>>>(wih, wihT, 512, 2048);
  conv_t_kernel<<<GRID1(1048576), blk, 0, stream>>>(whh, whhT, 512, 2048);
  conv_t_kernel<<<GRID1(262144), blk, 0, stream>>>(menc_w, mencT, 512, 512);
  conv_t_kernel<<<GRID1(131072), blk, 0, stream>>>(h1_w1, w1aT1, 512, 256);
  conv_t_kernel<<<GRID1(131072), blk, 0, stream>>>(h1_w1 + 512 * 256, w1bT1, 512, 256);
  conv_t_kernel<<<GRID1(131072), blk, 0, stream>>>(h2_w1, w1aT2, 512, 256);
  conv_t_kernel<<<GRID1(131072), blk, 0, stream>>>(h2_w1 + 512 * 256, w1bT2, 512, 256);
  conv_t_kernel<<<GRID1(16384), blk, 0, stream>>>(h1_w2, w2T1, 256, 64);
  conv_t_kernel<<<GRID1(16384), blk, 0, stream>>>(h2_w2, w2T2, 256, 64);
  conv_t_kernel<<<GRID1(262144), blk, 0, stream>>>(g1_W, g1WT, 512, 512);
  conv_t_kernel<<<GRID1(262144), blk, 0, stream>>>(g2_W, g2WT, 512, 512);
  conv_t_kernel<<<GRID1(262144), blk, 0, stream>>>(mdec_w, mdecT, 512, 512);
  conv_t_kernel<<<GRID1(20480), blk, 0, stream>>>(act_w, actWT, 1024, 20);

  // 1. x = obs @ enc_w + enc_b
  gemm_wmma<false, true><<<dim3(8, 4, 1), blk, 0, stream>>>(
      obs16, 128, 0, encT, 128, 0, enc_b, x_enc, 512, 0, 256, 512, 128);
  conv_kernel<<<GRID1(131072), blk, 0, stream>>>(x_enc, x16, 131072);
  // 2. gates = x@wih + bih; gates += hidden@whh + bhh
  gemm_wmma<false, true><<<dim3(32, 4, 1), blk, 0, stream>>>(
      x16, 512, 0, wihT, 512, 0, bih, gates, 2048, 0, 256, 2048, 512);
  gemm_wmma<true, true><<<dim3(32, 4, 1), blk, 0, stream>>>(
      hid16, 512, 0, whhT, 512, 0, bhh, gates, 2048, 0, 256, 2048, 512);
  // 3. LSTM pointwise
  lstm_kernel<<<512, blk, 0, stream>>>(gates, cell, out_h, out_c, h_new, h16);
  // 4. comm0 = h_new @ menc_w + menc_b
  gemm_wmma<false, true><<<dim3(8, 4, 1), blk, 0, stream>>>(
      h16, 512, 0, mencT, 512, 0, menc_b, comm0, 512, 0, 256, 512, 512);
  conv_kernel<<<GRID1(131072), blk, 0, stream>>>(comm0, c016, 131072);
  // 5. li/lj projections
  gemm_wmma<false, false><<<dim3(4, 4, 1), blk, 0, stream>>>(
      c016, 512, 0, w1aT1, 512, 0, nullptr, li1, 256, 0, 256, 256, 512);
  gemm_wmma<false, false><<<dim3(4, 4, 1), blk, 0, stream>>>(
      c016, 512, 0, w1bT1, 512, 0, nullptr, lj1, 256, 0, 256, 256, 512);
  gemm_wmma<false, false><<<dim3(4, 4, 1), blk, 0, stream>>>(
      c016, 512, 0, w1aT2, 512, 0, nullptr, li2, 256, 0, 256, 256, 512);
  gemm_wmma<false, false><<<dim3(4, 4, 1), blk, 0, stream>>>(
      c016, 512, 0, w1bT2, 512, 0, nullptr, lj2, 256, 0, 256, 256, 512);
  // 6. hard adjacencies (Gumbel argmax)
  hard_adj_kernel<<<dim3(32, 32), blk, 0, stream>>>(
      li1, lj1, h1_b1, w2T1, h1_b2, h1_w3, h1_b3, adj1, 0x1111u);
  hard_adj_kernel<<<dim3(32, 32), blk, 0, stream>>>(
      li2, lj2, h2_b1, w2T2, h2_b2, h2_w3, h2_b3, adj2, 0x2222u);
  // 7. GAT1
  gemm_wmma<false, false><<<dim3(8, 4, 1), blk, 0, stream>>>(
      c016, 512, 0, g1WT, 512, 0, nullptr, hproj1, 512, 0, 256, 512, 512);
  conv_t_kernel<<<GRID1(131072), blk, 0, stream>>>(hproj1, hp1T, 256, 512);
  gat_coef_kernel<<<256, 128, 0, stream>>>(hproj1, g1_ai, g1_aj, ci1, cj1, 4, 128);
  gat_att_kernel<<<dim3(256, 4), 256, 0, stream>>>(ci1, cj1, adj1, att116, 4);
  gemm_wmma<false, false><<<dim3(2, 4, 4), blk, 0, stream>>>(
      att116, NN, (long)NN * NN, hp1T, 256, 128 * 256, nullptr,
      g1o, 512, 128, 256, 128, 256);
  bias_elu_kernel<<<512, blk, 0, stream>>>(g1o, g1_b, comm2, c216);
  // 8. GAT2
  gemm_wmma<false, false><<<dim3(8, 4, 1), blk, 0, stream>>>(
      c216, 512, 0, g2WT, 512, 0, nullptr, hproj2, 512, 0, 256, 512, 512);
  conv_t_kernel<<<GRID1(131072), blk, 0, stream>>>(hproj2, hp2T, 256, 512);
  gat_coef_kernel<<<256, 128, 0, stream>>>(hproj2, g2_ai, g2_aj, ci2, cj2, 1, 512);
  gat_att_kernel<<<dim3(256, 1), 256, 0, stream>>>(ci2, cj2, adj2, att216, 1);
  gemm_wmma<false, false><<<dim3(8, 4, 1), blk, 0, stream>>>(
      att216, NN, 0, hp2T, 256, 0, nullptr, g2o, 512, 0, 256, 512, 256);
  bias_add_kernel<<<512, blk, 0, stream>>>(g2o, g2_b, comm3, c316);
  // 9. mdec
  gemm_wmma<false, true><<<dim3(8, 4, 1), blk, 0, stream>>>(
      c316, 512, 0, mdecT, 512, 0, mdec_b, comm4, 512, 0, 256, 512, 512);
  // 10. heads
  cat_kernel<<<1024, blk, 0, stream>>>(h_new, comm4, catb, cat16);
  value_kernel<<<256, 128, 0, stream>>>(catb, val_w, val_b, out_value);
  gemm_wmma<false, true><<<dim3(1, 4, 1), blk, 0, stream>>>(
      cat16, 1024, 0, actWT, 1024, 0, act_b, logits, 20, 0, 256, 20, 1024);
  logsoftmax_kernel<<<256, 32, 0, stream>>>(logits, out_action);
#undef GRID1
}
